// Pixel_Self_Attention_42339787604327
// MI455X (gfx1250) — matrix operand order repaired and checked by reading, not compiled, not run on hardware
//
#include <hip/hip_runtime.h>
#include <hip/hip_bf16.h>

typedef _Float16 __attribute__((ext_vector_type(8)))  v8h;
typedef _Float16 __attribute__((ext_vector_type(16))) v16h;
typedef float    __attribute__((ext_vector_type(8)))  v8f;

#define BATCH 2
#define CH    64
#define NPIX  9216          // 96*96
#define NT    576           // NPIX/16
#define WAVES 4
#define LOG2E 1.44269504088896340736f

// Build a 16-element f16 vector (8 VGPRs) from two 16B loads.
__device__ __forceinline__ v16h make_v16(v8h lo, v8h hi) {
    v16h r;
#pragma unroll
    for (int i = 0; i < 8; ++i) { r[i] = lo[i]; r[i + 8] = hi[i]; }
    return r;
}

// A fragment (16x32, M=pixels p0..p0+15, K=channels kb..kb+31) from qT[N][C].
// Contiguous k order used here: lane m = L&15, khalf = L>>4; v0-7 hold K = khalf*16 + 0..15
__device__ __forceinline__ v16h load_fragA_qT(const _Float16* qTb, int p0, int kb, int lane) {
    int m = lane & 15, h = lane >> 4;
    const _Float16* row = qTb + (size_t)(p0 + m) * CH;
    v8h lo = *reinterpret_cast<const v8h*>(row + kb + h * 16);
    v8h hi = *reinterpret_cast<const v8h*>(row + kb + h * 16 + 8);
    return make_v16(lo, hi);
}

// B fragment (32x16, K=channels kb..kb+31, N=pixels j0..j0+15) from qT[N][C].
// Contiguous k order used here: lane n = L&15, khalf = L>>4; v0-7 hold K = khalf*16 + 0..15
__device__ __forceinline__ v16h load_fragB_qT(const _Float16* qTb, int j0, int kb, int lane) {
    int n = lane & 15, h = lane >> 4;
    const _Float16* row = qTb + (size_t)(j0 + n) * CH;
    v8h lo = *reinterpret_cast<const v8h*>(row + kb + h * 16);
    v8h hi = *reinterpret_cast<const v8h*>(row + kb + h * 16 + 8);
    return make_v16(lo, hi);
}

// ---------------------------------------------------------------------------
// Kernel 0: convert x (f32 [b][c][i]) into two f16 layouts; thread 0 also
// zero-inits the maxnorm cell (stream ordering makes this safe).
// ---------------------------------------------------------------------------
__global__ void psa_convert(const float* __restrict__ x,
                            _Float16* __restrict__ qT,
                            _Float16* __restrict__ qC,
                            float* __restrict__ maxnorm) {
    int idx = blockIdx.x * blockDim.x + threadIdx.x;
    if (idx == 0) *maxnorm = 0.0f;
    if (idx >= BATCH * CH * NPIX) return;
    int b   = idx / (CH * NPIX);
    int rem = idx - b * CH * NPIX;
    int c   = rem / NPIX;
    int i   = rem - c * NPIX;
    _Float16 hv = (_Float16)x[idx];
    qC[idx] = hv;
    qT[((size_t)b * NPIX + i) * CH + c] = hv;
}

// ---------------------------------------------------------------------------
// Kernel 0b: per-row L2 norms of the f16 q rows + global max norm.
// M_i = norm_i * maxnorm is a Cauchy-Schwarz upper bound on row i of energy,
// used as the (exact, shift-invariant) softmax shift -> no max tracking.
// ---------------------------------------------------------------------------
__global__ __launch_bounds__(256) void psa_norms(const _Float16* __restrict__ qT,
                                                 float* __restrict__ rowNorm,
                                                 float* __restrict__ maxnorm) {
    __shared__ float sm[256];
    int tid = threadIdx.x;
    int i = blockIdx.x * 256 + tid;                 // BATCH*NPIX = 18432 rows
    const _Float16* row = qT + (size_t)i * CH;
    float ss = 0.f;
#pragma unroll
    for (int k = 0; k < CH; k += 8) {
        v8h v = *reinterpret_cast<const v8h*>(row + k);
#pragma unroll
        for (int j = 0; j < 8; ++j) { float f = (float)v[j]; ss = fmaf(f, f, ss); }
    }
    float nrm = sqrtf(ss);
    rowNorm[i] = nrm;
    sm[tid] = nrm;
    __syncthreads();
    for (int s2 = 128; s2 > 0; s2 >>= 1) {
        if (tid < s2) sm[tid] = fmaxf(sm[tid], sm[tid + s2]);
        __syncthreads();
    }
    // Non-negative floats compare like their int bit patterns.
    if (tid == 0) atomicMax(reinterpret_cast<int*>(maxnorm), __float_as_int(sm[0]));
}

// ---------------------------------------------------------------------------
// Kernel 1: per-row sum of exp(e - M_i).  Block = (b, i-tile), 4 waves.
// With the analytic shift the per-element work is one fma + one v_exp.
// Writes rowM = M_i (the shift) and rowIL = 1/sum for pass 2.
// ---------------------------------------------------------------------------
__global__ __launch_bounds__(128) void psa_rowstats(const _Float16* __restrict__ qT,
                                                    const float* __restrict__ rowNorm,
                                                    const float* __restrict__ maxnorm,
                                                    float* __restrict__ rowM,
                                                    float* __restrict__ rowIL) {
    __shared__ float sS[WAVES][16];
    int blk = blockIdx.x;
    int b = blk / NT, it = blk % NT;
    int i0 = it * 16;
    const _Float16* qTb = qT + (size_t)b * NPIX * CH;
    const float* rN = rowNorm + (size_t)b * NPIX;
    int lane = threadIdx.x & 31, w = threadIdx.x >> 5;
    int h = lane >> 4;
    float mn = *maxnorm;

    v16h a0 = load_fragA_qT(qTb, i0, 0, lane);
    v16h a1 = load_fragA_qT(qTb, i0, 32, lane);

    // Rows handled by this lane: i0 + 8*h + r, r = 0..7 (consecutive).
    int mbase = i0 + 8 * h;
    float4 n0 = *reinterpret_cast<const float4*>(rN + mbase);
    float4 n1 = *reinterpret_cast<const float4*>(rN + mbase + 4);
    float nr[8] = {n0.x, n0.y, n0.z, n0.w, n1.x, n1.y, n1.z, n1.w};
    float nML[8];                                   // -M_i * log2(e), folded into exp2
#pragma unroll
    for (int r = 0; r < 8; ++r) nML[r] = -(nr[r] * mn) * LOG2E;

    float runS[8];
#pragma unroll
    for (int r = 0; r < 8; ++r) runS[r] = 0.f;

    for (int jt = w; jt < NT; jt += WAVES) {        // uniform: 576/4 per wave
        int j0 = jt * 16;
        v16h b0 = load_fragB_qT(qTb, j0, 0, lane);
        v16h b1 = load_fragB_qT(qTb, j0, 32, lane);
        v8f e = {};
        e = __builtin_amdgcn_wmma_f32_16x16x32_f16(false, a0, false, b0, (short)0, e, false, false);
        e = __builtin_amdgcn_wmma_f32_16x16x32_f16(false, a1, false, b1, (short)0, e, false, false);
#pragma unroll
        for (int r = 0; r < 8; ++r)                 // one fma + one exp per element
            runS[r] += __builtin_amdgcn_exp2f(fmaf(e[r], LOG2E, nML[r]));
    }
    // Sum across the 16 lanes of each half (xor masks 1..8 stay within half).
#pragma unroll
    for (int r = 0; r < 8; ++r) {
        float s = runS[r];
#pragma unroll
        for (int mask = 1; mask < 16; mask <<= 1)
            s += __shfl_xor(s, mask, 32);
        runS[r] = s;
    }
    if ((lane & 15) == 0) {                         // lanes 0 and 16 publish
#pragma unroll
        for (int r = 0; r < 8; ++r) sS[w][h * 8 + r] = runS[r];
    }
    __syncthreads();
    if (threadIdx.x < 16) {                         // combine the 4 waves
        float s = 0.f;
        for (int q = 0; q < WAVES; ++q) s += sS[q][threadIdx.x];
        size_t gi = (size_t)b * NPIX + i0 + threadIdx.x;
        rowM [gi] = rN[i0 + threadIdx.x] * mn;      // the shift used
        rowIL[gi] = 1.0f / s;
    }
}

// ---------------------------------------------------------------------------
// Kernel 2: out[c][j] = sum_i P[i][j] q[c][i], P = exp(E - M_i) * il_i, fused
// epilogue gamma*out + x.  Block = (b, j-tile), 4 waves split the i range.
// P staged in LDS as packed f16, transposed [n][k], padded rows.
// ---------------------------------------------------------------------------
#define PSTRIDE 40   // 32 halves + 8 pad -> 80B row stride, spreads LDS banks

__global__ __launch_bounds__(128) void psa_output(const _Float16* __restrict__ qT,
                                                  const _Float16* __restrict__ qC,
                                                  const float* __restrict__ rowM,
                                                  const float* __restrict__ rowIL,
                                                  const float* __restrict__ x,
                                                  const float* __restrict__ gamma,
                                                  float* __restrict__ out) {
    __shared__ _Float16 PwT[WAVES][16][PSTRIDE];  // per-wave P tile, f16, [n][k]
    __shared__ float accb[WAVES][4][16][16];      // per-wave partial D tiles
    int blk = blockIdx.x;
    int b = blk / NT, jt = blk % NT;
    int j0 = jt * 16;
    const _Float16* qTb = qT + (size_t)b * NPIX * CH;
    const _Float16* qCb = qC + (size_t)b * CH * NPIX;
    const float* rM  = rowM  + (size_t)b * NPIX;
    const float* rIL = rowIL + (size_t)b * NPIX;
    int lane = threadIdx.x & 31, w = threadIdx.x >> 5;
    int n = lane & 15, h = lane >> 4;

    // Energy B fragments (j fixed for whole block) hoisted out of the loop.
    v16h be0 = load_fragB_qT(qTb, j0, 0, lane);
    v16h be1 = load_fragB_qT(qTb, j0, 32, lane);

    v8f acc[4] = {};                          // c-subtiles 0..3 (c = cs*16+m)

    const int NCHUNK = NPIX / 32;             // 288, 72 per wave: uniform
    for (int ic = w; ic < NCHUNK; ic += WAVES) {
        int i0 = ic * 32;
        // ---- recompute energy for 32 i-rows, write f16 P tile to LDS ----
#pragma unroll
        for (int t = 0; t < 2; ++t) {
            int it0 = i0 + t * 16;
            v16h ae0 = load_fragA_qT(qTb, it0, 0, lane);
            v16h ae1 = load_fragA_qT(qTb, it0, 32, lane);
            v8f e = {};
            e = __builtin_amdgcn_wmma_f32_16x16x32_f16(false, ae0, false, be0, (short)0, e, false, false);
            e = __builtin_amdgcn_wmma_f32_16x16x32_f16(false, ae1, false, be1, (short)0, e, false, false);
            int mbase = it0 + 8 * h;          // D layout: row = r + 8*h (consecutive)
            float4 m0 = *reinterpret_cast<const float4*>(rM  + mbase);
            float4 m1 = *reinterpret_cast<const float4*>(rM  + mbase + 4);
            float4 l0 = *reinterpret_cast<const float4*>(rIL + mbase);
            float4 l1 = *reinterpret_cast<const float4*>(rIL + mbase + 4);
            float mm[8] = {m0.x, m0.y, m0.z, m0.w, m1.x, m1.y, m1.z, m1.w};
            float ll[8] = {l0.x, l0.y, l0.z, l0.w, l1.x, l1.y, l1.z, l1.w};
            v8h ph;
#pragma unroll
            for (int r = 0; r < 8; ++r) {
                float p = __builtin_amdgcn_exp2f(fmaf(e[r], LOG2E, -mm[r] * LOG2E)) * ll[r];
                ph[r] = (_Float16)p;
            }
            // k = t*16 + 8*h + r, r consecutive -> one 16B LDS store
            *reinterpret_cast<v8h*>(&PwT[w][n][t * 16 + 8 * h]) = ph;
        }
        // ---- B fragment: lane n fixed, K = h*16 + 0..15 contiguous in LDS ----
        v8h plo = *reinterpret_cast<const v8h*>(&PwT[w][n][h * 16]);
        v8h phi = *reinterpret_cast<const v8h*>(&PwT[w][n][h * 16 + 8]);
        v16h bp = make_v16(plo, phi);
        // ---- 4 c-subtiles: D += q(16x32) * P(32x16) ----
#pragma unroll
        for (int cs = 0; cs < 4; ++cs) {
            const _Float16* row = qCb + (size_t)(cs * 16 + n) * NPIX;  // A lane row = channel
            v8h lo = *reinterpret_cast<const v8h*>(row + i0 + h * 16);
            v8h hi = *reinterpret_cast<const v8h*>(row + i0 + h * 16 + 8);
            v16h ao = make_v16(lo, hi);
            acc[cs] = __builtin_amdgcn_wmma_f32_16x16x32_f16(false, ao, false, bp, (short)0, acc[cs], false, false);
        }
    }
    __syncthreads();                          // all waves done with PwT
#pragma unroll
    for (int cs = 0; cs < 4; ++cs)
#pragma unroll
        for (int r = 0; r < 8; ++r)
            accb[w][cs][r + 8 * h][n] = acc[cs][r];
    __syncthreads();
    // Combine 4 waves + fused epilogue: 1024 outputs, 8 per thread.
    float g = gamma[0];
    for (int e3 = threadIdx.x; e3 < 1024; e3 += 128) {
        int cs = e3 >> 8;
        int m  = (e3 >> 4) & 15;
        int nn = e3 & 15;
        float s = accb[0][cs][m][nn] + accb[1][cs][m][nn]
                + accb[2][cs][m][nn] + accb[3][cs][m][nn];
        int c = cs * 16 + m;
        size_t idx = ((size_t)b * CH + c) * NPIX + j0 + nn;
        out[idx] = g * s + x[idx];
    }
}

// ---------------------------------------------------------------------------
extern "C" void kernel_launch(void* const* d_in, const int* in_sizes, int n_in,
                              void* d_out, int out_size, void* d_ws, size_t ws_size,
                              hipStream_t stream) {
    const float* x     = (const float*)d_in[0];
    const float* gamma = (const float*)d_in[1];
    float* out = (float*)d_out;

    char* ws = (char*)d_ws;
    size_t qbytes = (size_t)BATCH * NPIX * CH * sizeof(_Float16);   // 2.36 MB each
    _Float16* qT = (_Float16*)ws;                                   // [b][i][c]
    _Float16* qC = (_Float16*)(ws + qbytes);                        // [b][c][i]
    float* rowM   = (float*)(ws + 2 * qbytes);
    float* rowIL  = rowM   + (size_t)BATCH * NPIX;
    float* rowNrm = rowIL  + (size_t)BATCH * NPIX;
    float* maxnrm = rowNrm + (size_t)BATCH * NPIX;                  // 1 float

    int total = BATCH * CH * NPIX;
    psa_convert<<<(total + 255) / 256, 256, 0, stream>>>(x, qT, qC, maxnrm);
    psa_norms<<<(BATCH * NPIX) / 256, 256, 0, stream>>>(qT, rowNrm, maxnrm);
    psa_rowstats<<<BATCH * NT, 128, 0, stream>>>(qT, rowNrm, maxnrm, rowM, rowIL);
    psa_output<<<BATCH * NT, 128, 0, stream>>>(qT, qC, rowM, rowIL, x, gamma, out);
}
